// WorldModel_65377992180400
// MI455X (gfx1250) — matrix operand order repaired and checked by reading, not compiled, not run on hardware
//
#include <hip/hip_runtime.h>
#include <hip/hip_bf16.h>

// ---------------------------------------------------------------------------
// ACT-style halting MLP, fully fused for MI455X (gfx1250).
//   - 9 steps x (sigmoid gate, relu(128->256), relu(256->256), tanh(256->128),
//     gated state blend) in one kernel; state never leaves LDS (~309 GFLOP vs
//     128MB of mandatory HBM traffic -> heavily compute-bound -> f16 WMMA
//     with f32 accumulate is the right precision/path).
//   - Weights pre-converted to f16 in d_ws by a prep kernel, then staged into
//     LDS via global_load_async_to_lds_b128 (ASYNCcnt) once per block.
//   - Block = 128 threads = 4 wave32; each wave owns a 16-row batch tile.
//   - Two independent WMMA accumulator chains per tile loop to keep the
//     matrix pipe busy at 1 wave/SIMD occupancy (LDS-limited: ~311KB/WGP).
// ---------------------------------------------------------------------------

typedef __attribute__((ext_vector_type(16))) _Float16 v16h;
typedef __attribute__((ext_vector_type(8)))  _Float16 v8h;
typedef __attribute__((ext_vector_type(4)))  _Float16 v4h;
typedef __attribute__((ext_vector_type(8)))  float    v8f;

// Pointer types matching the async-LDS builtin signature (from the hipcc
// diagnostic: param is "__attribute__((vector_size(16))) int AS1 *").
typedef int v4i_vs __attribute__((vector_size(16)));
typedef __attribute__((address_space(1))) v4i_vs* gptr_b128;
typedef __attribute__((address_space(3))) v4i_vs* lptr_b128;

#define D        128
#define D2       256
#define WAVES    4
#define RPW      16                 // rows per wave
#define ROWS     (WAVES * RPW)      // 64 rows per block
#define NSTEPS   9                  // MAX_CYCLES + 1
#define WHALFS   (D2*D + D2*D2 + D*D2)   // 131072 f16 weight elements
#define WBYTES   (WHALFS * 2)            // 262144 bytes

// ---- LDS layout (bytes); W1|W2|W3 contiguous so one async copy fills them --
#define OFF_W1   0                                  // 256x128 f16 = 64KB
#define OFF_W2   (OFF_W1 + D2 * D  * 2)             // 256x256 f16 = 128KB
#define OFF_W3   (OFF_W2 + D2 * D2 * 2)             // 128x256 f16 = 64KB
#define OFF_ST   (OFF_W3 + D  * D2 * 2)             // 64x128 f16 state
#define OFF_H1   (OFF_ST + ROWS * D  * 2)           // 64x256 f16 hidden1
#define OFF_SCR  (OFF_H1 + ROWS * D2 * 2)           // 4 x (16x32) f16 scratch
#define OFF_B1   (OFF_SCR + WAVES * RPW * 32 * 2)
#define OFF_B2   (OFF_B1 + D2 * 4)
#define OFF_B3   (OFF_B2 + D2 * 4)
#define OFF_WH   (OFF_B3 + D * 4)
#define OFF_P    (OFF_WH + D * 4)
#define OFF_ACT  (OFF_P + ROWS * 4)
#define OFF_BH   (OFF_ACT + ROWS * 4)
#define LDS_BYTES (OFF_BH + 4)                      // ~311KB < 320KB/WGP

#if __has_builtin(__builtin_amdgcn_global_load_async_to_lds_b128)
#define HAVE_ASYNC_LDS 1
#endif

#define WMMA_F16(A, B, C) \
  __builtin_amdgcn_wmma_f32_16x16x32_f16(false, (A), false, (B), (short)0, (C), false, false)

// A fragment (16x32, f16), ISA layout: lane L holds row M=L&15;
// halfs 0..7 -> K = 8*(L>>4)+i, halfs 8..15 -> K = 16 + 8*(L>>4)+i.
__device__ __forceinline__ v16h load_a_frag(const _Float16* base, int lane, int stride) {
  const _Float16* p = base + (lane & 15) * stride + ((lane >> 4) << 3);
  v8h lo = *(const v8h*)(p);
  v8h hi = *(const v8h*)(p + 16);
  return __builtin_shufflevector(lo, hi, 0,1,2,3,4,5,6,7,8,9,10,11,12,13,14,15);
}

// B fragment (32x16, f16), ISA layout: lane L holds col N=L&15;
// halfs 0..7 -> K = 8*(L>>4)+i, halfs 8..15 -> K = 16 + 8*(L>>4)+i. Weight stored [N][K] row-major.
__device__ __forceinline__ v16h load_b_frag(const _Float16* base, int lane, int stride) {
  const _Float16* p = base + (lane & 15) * stride + ((lane >> 4) << 3);
  v8h lo = *(const v8h*)(p);
  v8h hi = *(const v8h*)(p + 16);
  return __builtin_shufflevector(lo, hi, 0,1,2,3,4,5,6,7,8,9,10,11,12,13,14,15);
}

// D tile (f32) + bias -> relu -> f16 store; lane holds col=colL, rows rhi..rhi+7.
__device__ __forceinline__ void store_relu_f16(_Float16* dst, int stride, v8f acc,
                                               float bias, int rhi, int colL) {
  #pragma unroll
  for (int r = 0; r < 8; ++r) {
    float v = acc[r] + bias;
    v = v > 0.f ? v : 0.f;
    dst[(rhi + r) * stride + colL] = (_Float16)v;
  }
}

// f32 -> f16 weight pre-conversion into workspace (once per launch, tiny).
__global__ void prep_weights(const float* __restrict__ W1, const float* __restrict__ W2,
                             const float* __restrict__ W3, _Float16* __restrict__ ws) {
  int i = blockIdx.x * 256 + threadIdx.x;
  if (i < D2 * D)                 ws[i] = (_Float16)W1[i];
  else if (i < D2 * D + D2 * D2)  ws[i] = (_Float16)W2[i - D2 * D];
  else if (i < WHALFS)            ws[i] = (_Float16)W3[i - D2 * D - D2 * D2];
}

__global__ __launch_bounds__(128) void act_fused_kernel(
    const float* __restrict__ gX,  const float* __restrict__ gW1, const float* __restrict__ gB1,
    const float* __restrict__ gW2, const float* __restrict__ gB2, const float* __restrict__ gW3,
    const float* __restrict__ gB3, const float* __restrict__ gWh, const float* __restrict__ gBh,
    float* __restrict__ gOut, const _Float16* __restrict__ gWs, int useWs)
{
  extern __shared__ char smem[];
  _Float16* sW1 = (_Float16*)(smem + OFF_W1);
  _Float16* sW2 = (_Float16*)(smem + OFF_W2);
  _Float16* sW3 = (_Float16*)(smem + OFF_W3);
  _Float16* sST = (_Float16*)(smem + OFF_ST);
  _Float16* sH1 = (_Float16*)(smem + OFF_H1);
  _Float16* sSCR = (_Float16*)(smem + OFF_SCR);
  float* sB1 = (float*)(smem + OFF_B1);
  float* sB2 = (float*)(smem + OFF_B2);
  float* sB3 = (float*)(smem + OFF_B3);
  float* sWh = (float*)(smem + OFF_WH);
  float* sP  = (float*)(smem + OFF_P);
  int*   sAct = (int*)(smem + OFF_ACT);
  float* sBH = (float*)(smem + OFF_BH);

  const int tid  = threadIdx.x;
  const int wv   = tid >> 5;
  const int lane = tid & 31;
  const int colL = lane & 15;
  const int rhi  = (lane >> 4) << 3;          // row offset 0 or 8 in C/D tiles
  const size_t row0 = (size_t)blockIdx.x * ROWS;

  // ---- stage weights into LDS ---------------------------------------------
  if (useWs) {
#ifdef HAVE_ASYNC_LDS
    // DMA-style: f16 weights -> LDS, tracked by ASYNCcnt (no VGPR round-trip)
    const char* gw = (const char*)gWs;
    char* lw = smem + OFF_W1;
    for (int i = tid * 16; i < WBYTES; i += 128 * 16)
      __builtin_amdgcn_global_load_async_to_lds_b128(
          (gptr_b128)(uintptr_t)(gw + i),
          (lptr_b128)(unsigned)(uintptr_t)(lw + i),
          0, 0);
#else
    for (int i = tid * 8; i < WHALFS; i += 128 * 8)
      *(v8h*)(sW1 + i) = *(const v8h*)(gWs + i);
#endif
  } else {
    for (int i = tid; i < D2 * D;  i += 128) sW1[i] = (_Float16)gW1[i];
    for (int i = tid; i < D2 * D2; i += 128) sW2[i] = (_Float16)gW2[i];
    for (int i = tid; i < D * D2;  i += 128) sW3[i] = (_Float16)gW3[i];
  }
  for (int i = tid; i < D2; i += 128) { sB1[i] = gB1[i]; sB2[i] = gB2[i]; }
  if (tid < D) { sB3[tid] = gB3[tid]; sWh[tid] = gWh[tid]; }
  if (tid == 0) sBH[0] = gBh[0];
  for (int i = 4 * tid; i < ROWS * D; i += 4 * 128) {
    float4 xv = *(const float4*)(gX + row0 * D + i);
    v4h h = { (_Float16)xv.x, (_Float16)xv.y, (_Float16)xv.z, (_Float16)xv.w };
    *(v4h*)(sST + i) = h;
  }
  if (tid < ROWS) sAct[tid] = 1;
#ifdef HAVE_ASYNC_LDS
  if (useWs) {
#if __has_builtin(__builtin_amdgcn_s_wait_asynccnt)
    __builtin_amdgcn_s_wait_asynccnt(0);
#else
    asm volatile("s_wait_asynccnt 0" ::: "memory");
#endif
  }
#endif
  __syncthreads();

  _Float16* stateW = sST  + wv * RPW * D;     // this wave's 16x128 state tile
  _Float16* h1W    = sH1  + wv * RPW * D2;    // this wave's 16x256 hidden tile
  _Float16* scrW   = sSCR + wv * RPW * 32;    // this wave's 16x32 scratch tile

  for (int step = 0; step < NSTEPS; ++step) {
    // ---- halting probability p = sigmoid(state . Wh + bh), cont mask ------
    if (tid < ROWS) {
      float s = 0.f;
      const _Float16* sr = sST + tid * D;
      #pragma unroll
      for (int k = 0; k < D; k += 8) {
        v8h h = *(const v8h*)(sr + k);
        #pragma unroll
        for (int j = 0; j < 8; ++j) s += (float)h[j] * sWh[k + j];
      }
      float pv = 1.f / (1.f + expf(-(s + sBH[0])));
      sP[tid]   = pv;
      sAct[tid] = sAct[tid] & ((pv <= 0.5f) ? 1 : 0);
    }
    __syncthreads();

    // ---- layer 1: h1 = relu(state @ W1^T + b1)   (16x128 @ 128x256) -------
    // two independent accumulator chains per iteration
    v16h a1[4];
    #pragma unroll
    for (int kt = 0; kt < 4; ++kt) a1[kt] = load_a_frag(stateW + kt * 32, lane, D);
    for (int nt = 0; nt < 16; nt += 2) {
      v8f acc0 = {}, acc1 = {};
      #pragma unroll
      for (int kt = 0; kt < 4; ++kt) {
        v16h b0 = load_b_frag(sW1 + (nt + 0) * 16 * D + kt * 32, lane, D);
        v16h b1 = load_b_frag(sW1 + (nt + 1) * 16 * D + kt * 32, lane, D);
        acc0 = WMMA_F16(a1[kt], b0, acc0);
        acc1 = WMMA_F16(a1[kt], b1, acc1);
      }
      store_relu_f16(h1W + (nt + 0) * 16, D2, acc0, sB1[(nt + 0) * 16 + colL], rhi, colL);
      store_relu_f16(h1W + (nt + 1) * 16, D2, acc1, sB1[(nt + 1) * 16 + colL], rhi, colL);
    }
    __builtin_amdgcn_wave_barrier();          // h1 is per-wave; DS ops in-order

    // ---- layer 2 (relu, 256->256) fused into layer 3 accumulation --------
    v16h aH[8];
    #pragma unroll
    for (int kt = 0; kt < 8; ++kt) aH[kt] = load_a_frag(h1W + kt * 32, lane, D2);
    v8f accC[8] = {};                         // layer-3 accumulators (16x128)
    for (int j = 0; j < 8; ++j) {             // produce h2 cols 32j..32j+31
      v8f acc0 = {}, acc1 = {};
      #pragma unroll
      for (int kt = 0; kt < 8; ++kt) {
        v16h b0 = load_b_frag(sW2 + (2 * j + 0) * 16 * D2 + kt * 32, lane, D2);
        v16h b1 = load_b_frag(sW2 + (2 * j + 1) * 16 * D2 + kt * 32, lane, D2);
        acc0 = WMMA_F16(aH[kt], b0, acc0);
        acc1 = WMMA_F16(aH[kt], b1, acc1);
      }
      store_relu_f16(scrW,      32, acc0, sB2[(2 * j + 0) * 16 + colL], rhi, colL);
      store_relu_f16(scrW + 16, 32, acc1, sB2[(2 * j + 1) * 16 + colL], rhi, colL);
      __builtin_amdgcn_wave_barrier();        // scratch is per-wave
      v16h a3 = load_a_frag(scrW, lane, 32);  // D-layout -> A-layout relayout
      #pragma unroll
      for (int n3 = 0; n3 < 8; ++n3) {
        v16h b = load_b_frag(sW3 + n3 * 16 * D2 + j * 32, lane, D2);
        accC[n3] = WMMA_F16(a3, b, accC[n3]);
      }
      __builtin_amdgcn_wave_barrier();        // before scratch is overwritten
    }

    // ---- c = tanh(. + b3); state = cont ? c*p + (1-p)*state : state -------
    #pragma unroll
    for (int n3 = 0; n3 < 8; ++n3) {
      const int col = n3 * 16 + colL;
      float b3v = sB3[col];
      #pragma unroll
      for (int r = 0; r < 8; ++r) {
        const int grow = wv * RPW + rhi + r;  // own wave's rows only
        float c  = tanhf(accC[n3][r] + b3v);
        float pv = sP[grow];
        float old = (float)sST[grow * D + col];
        float nv = sAct[grow] ? (c * pv + (1.f - pv) * old) : old;
        sST[grow * D + col] = (_Float16)nv;
      }
    }
    __syncthreads();                          // p-phase reads all rows next
  }

  // ---- write final state (f32) --------------------------------------------
  for (int i = 4 * tid; i < ROWS * D; i += 4 * 128) {
    v4h h = *(const v4h*)(sST + i);
    float4 o = make_float4((float)h[0], (float)h[1], (float)h[2], (float)h[3]);
    *(float4*)(gOut + row0 * D + i) = o;
  }
}

extern "C" void kernel_launch(void* const* d_in, const int* in_sizes, int n_in,
                              void* d_out, int out_size, void* d_ws, size_t ws_size,
                              hipStream_t stream) {
  (void)n_in; (void)out_size;
  const float* x  = (const float*)d_in[0];
  const float* W1 = (const float*)d_in[1];
  const float* b1 = (const float*)d_in[2];
  const float* W2 = (const float*)d_in[3];
  const float* b2 = (const float*)d_in[4];
  const float* W3 = (const float*)d_in[5];
  const float* b3 = (const float*)d_in[6];
  const float* Wh = (const float*)d_in[7];
  const float* bh = (const float*)d_in[8];
  float* out = (float*)d_out;

  const int batch = in_sizes[0] / D;          // 131072
  const int nblocks = batch / ROWS;           // 2048
  const int useWs = (d_ws != nullptr && ws_size >= (size_t)WBYTES) ? 1 : 0;

  if (useWs)
    prep_weights<<<dim3((WHALFS + 255) / 256), dim3(256), 0, stream>>>(
        W1, W2, W3, (_Float16*)d_ws);

  act_fused_kernel<<<dim3(nblocks), dim3(128), LDS_BYTES, stream>>>(
      x, W1, b1, W2, b2, W3, b3, Wh, bh, out, (const _Float16*)d_ws, useWs);
}
